// DeltaNet_31877247271443
// MI455X (gfx1250) — matrix operand order repaired and checked by reading, not compiled, not run on hardware
//
#include <hip/hip_runtime.h>
#include <hip/hip_bf16.h>
#include <math.h>

// ---------------- problem constants ----------------
#define BB 2
#define LL 2048
#define DD 1024
#define HH 4
#define DK 256
#define DVv 256
#define DVH 128
#define PP 5
#define CHK 32
#define NCH (LL/CHK)
#define BL (BB*LL)      // 4096
#define HF 2048
#define NSTAT 80

typedef __bf16 bf16;
typedef __attribute__((ext_vector_type(16))) __bf16 v16bf;
typedef __attribute__((ext_vector_type(8)))  float  v8f;

__device__ __forceinline__ bf16 f2bf(float f){
  unsigned u = __builtin_bit_cast(unsigned, f);
  u += 0x7fffu + ((u >> 16) & 1u);               // round-to-nearest-even
  unsigned short s = (unsigned short)(u >> 16);
  return __builtin_bit_cast(bf16, s);
}

__device__ __forceinline__ float wred32(float s){
  #pragma unroll
  for(int o = 16; o > 0; o >>= 1) s += __shfl_xor(s, o, 32);
  return s;
}

__device__ __forceinline__ v8f wmma_bf(v16bf a, v16bf b, v8f c){
  return __builtin_amdgcn_wmma_f32_16x16x32_bf16(false, a, false, b, (short)0, c, false, false);
}

// ---- fragment gathers for wave32 WMMA, contiguous k order (k = 16*half + j) in both A and B ----
// A[m][k] from row-major [rows][ld]; lanes 0-15: M=lane, lanes 16-31: M=lane-16
// vec[j] -> K = k0 + 16*half + j
__device__ __forceinline__ v16bf frag_a(const bf16* p, int ld, int m0, int k0, int lane){
  int half = lane >> 4;
  const bf16* row = p + (m0 + (lane & 15)) * ld + k0 + 16 * half;
  v16bf a;
  #pragma unroll
  for(int j = 0; j < 8; j++){ a[j] = row[j]; a[8 + j] = row[8 + j]; }
  return a;
}
// A[m][k] where A = srcT  (src stored [K rows][ld], element src[k][m])
__device__ __forceinline__ v16bf frag_aT(const bf16* p, int ld, int m0, int k0, int lane){
  int half = lane >> 4, m = m0 + (lane & 15);
  v16bf a;
  #pragma unroll
  for(int j = 0; j < 8; j++){
    a[j]     = p[(k0 + 16 * half + j) * ld + m];
    a[8 + j] = p[(k0 + 16 * half + 8 + j) * ld + m];
  }
  return a;
}
// B[k][n] from row-major [K][ld]; lane n = n0+(lane&15); vec[j] -> K = k0 + 16*half + j
__device__ __forceinline__ v16bf frag_b(const bf16* p, int ld, int k0, int n0, int lane){
  int half = lane >> 4;
  const bf16* col = p + (k0 + 16 * half) * ld + n0 + (lane & 15);
  v16bf b;
  #pragma unroll
  for(int j = 0; j < 16; j++) b[j] = col[j * ld];
  return b;
}
// B[k][n] from n-major (transposed) storage [N rows][ldk] -> fully contiguous gather
__device__ __forceinline__ v16bf frag_bT(const bf16* p, int ldk, int n0, int k0, int lane){
  int half = lane >> 4;
  const bf16* row = p + (n0 + (lane & 15)) * ldk + k0 + 16 * half;
  v16bf b;
  #pragma unroll
  for(int j = 0; j < 16; j++) b[j] = row[j];
  return b;
}
// B[k][n] from f32 row-major [K][ld], converted on the fly
__device__ __forceinline__ v16bf frag_b_f32(const float* p, int ld, int k0, int n0, int lane){
  int half = lane >> 4;
  const float* col = p + (k0 + 16 * half) * ld + n0 + (lane & 15);
  v16bf b;
  #pragma unroll
  for(int j = 0; j < 16; j++) b[j] = f2bf(col[j * ld]);
  return b;
}

// ---------------- generic bf16-WMMA GEMM: C[M,N] = A[M,K] @ W[K,N] ----------------
// tiles: block 64x128, 8 waves, each wave one 16x64 strip (4 wmma tiles)
__launch_bounds__(256)
__global__ void gemm_wmma(const float* __restrict__ A, const float* __restrict__ Bw,
                          float* __restrict__ C, int M, int N, int K){
  __shared__ bf16 As[64 * 32];     // [64][32] row major
  __shared__ bf16 Bs[128 * 32];    // [128][32] = W-tile transposed (n-major)
  const int tid = threadIdx.x, lane = tid & 31, wave = tid >> 5;
  const int bm = blockIdx.y * 64, bn = blockIdx.x * 128;
  const int m0 = (wave & 3) * 16, n0 = (wave >> 2) * 64;
  v8f acc[4] = {};
  for(int k0 = 0; k0 < K; k0 += 32){
    if(k0 + 32 < K){
      __builtin_prefetch(&A[(size_t)(bm + (tid >> 2)) * K + k0 + 32], 0, 0);
      __builtin_prefetch(&Bw[(size_t)(k0 + 32 + (tid >> 7)) * N + bn + (tid & 127)], 0, 0);
    }
    #pragma unroll
    for(int j = 0; j < 8; j++){
      int i = tid + 256 * j;
      As[i] = f2bf(A[(size_t)(bm + (i >> 5)) * K + k0 + (i & 31)]);
    }
    #pragma unroll
    for(int j = 0; j < 16; j++){
      int i = tid + 256 * j; int cc = i >> 7, n = i & 127;
      Bs[n * 32 + cc] = f2bf(Bw[(size_t)(k0 + cc) * N + bn + n]);
    }
    __syncthreads();
    v16bf a = frag_a(As, 32, m0, 0, lane);
    #pragma unroll
    for(int t = 0; t < 4; t++){
      v16bf b = frag_bT(Bs, 32, n0 + t * 16, 0, lane);
      acc[t] = wmma_bf(a, b, acc[t]);
    }
    __syncthreads();
  }
  const int half = lane >> 4, nn = lane & 15;
  #pragma unroll
  for(int t = 0; t < 4; t++)
    #pragma unroll
    for(int r = 0; r < 8; r++)
      C[(size_t)(bm + m0 + r + 8 * half) * N + bn + n0 + t * 16 + nn] = acc[t][r];
}

// ---------------- causal depthwise conv (+ optional SiLU) ----------------
__global__ void dwconv_k(const float* __restrict__ x, const float* __restrict__ w,
                         float* __restrict__ y, int Kc, int act){
  const long total = (long)BB * LL * DD;
  for(long idx = (long)blockIdx.x * 256 + threadIdx.x; idx < total; idx += (long)gridDim.x * 256){
    int c = (int)(idx % DD);
    long bl = idx / DD;
    int l = (int)(bl % LL);
    float acc = 0.f;
    for(int t = 0; t < Kc; t++){
      int li = l - (Kc - 1) + t;
      if(li >= 0) acc += w[c * Kc + t] * x[(bl - l + li) * DD + c];
    }
    if(act) acc = acc / (1.f + expf(-acc));   // silu
    y[idx] = acc;
  }
}

// ---------------- beta = sigmoid(hs @ Wb) ----------------
__global__ void beta_k(const float* __restrict__ hs, const float* __restrict__ Wb,
                       float* __restrict__ beta){
  int idx = blockIdx.x * 256 + threadIdx.x;
  if(idx >= BL * HH) return;
  int h = idx & 3, bl = idx >> 2;
  float s = 0.f;
  for(int d = 0; d < DD; d++) s += hs[(size_t)bl * DD + d] * Wb[d * HH + h];
  beta[idx] = 1.f / (1.f + expf(-s));
}

// ---------------- gW1_red[g][n] = sum_d gW1[1024 + g*256 + d][n] ----------------
__global__ void reduce_gw1_k(const float* __restrict__ gW1, float* __restrict__ g1red){
  int g = blockIdx.x >> 3;
  int col = ((blockIdx.x & 7) << 8) + threadIdx.x;
  const float* src = gW1 + (size_t)(DD + g * DVv) * HF + col;
  float s = 0.f;
  for(int d = 0; d < DVv; d++) s += src[(size_t)d * HF];
  g1red[(size_t)g * HF + col] = s;
}

// ---------------- per-(b,l) path stats ----------------
__global__ void stats_k(const float* __restrict__ p0, const float* __restrict__ p1,
                        const float* __restrict__ p2, const float* __restrict__ p3,
                        const float* __restrict__ p4, float* __restrict__ stat){
  int bl = blockIdx.x;
  int lane = threadIdx.x & 31, wave = threadIdx.x >> 5;
  for(int g = wave; g < 20; g += 8){
    int p = g >> 2, h = g & 3;
    const float* pp = (p == 0) ? p0 : (p == 1) ? p1 : (p == 2) ? p2 : (p == 3) ? p3 : p4;
    const float* base = pp + (size_t)bl * DD + h * DVv;
    float s = 0.f, s2 = 0.f, sa = 0.f;
    #pragma unroll
    for(int j = 0; j < 8; j++){
      float x = base[lane + 32 * j];
      s += x; s2 += x * x; sa += fabsf(x);
    }
    s = wred32(s); s2 = wred32(s2); sa = wred32(sa);
    if(lane == 0){
      float* o = stat + (size_t)bl * NSTAT + g * 4;
      o[0] = s * (1.f / DVv); o[1] = s2 * (1.f / DVv);
      o[2] = sa * (1.f / DVv); o[3] = sqrtf(s2);
    }
  }
}

// ---------------- gate MLP + softmax + path mix + RMSNorm ----------------
__launch_bounds__(256)
__global__ void gate_k(const float* __restrict__ g1pre, const float* __restrict__ stat,
                       const float* __restrict__ g1red, const float* __restrict__ gb1,
                       const float* __restrict__ gW2, const float* __restrict__ gb2,
                       const float* __restrict__ log_tau, const float* __restrict__ eps_logit,
                       const float* __restrict__ norm_w,
                       const float* __restrict__ p0, const float* __restrict__ p1,
                       const float* __restrict__ p2, const float* __restrict__ p3,
                       const float* __restrict__ p4, float* __restrict__ orb){
  __shared__ float h1[HF];
  __shared__ float statv[NSTAT];
  __shared__ float red[20 * 8];
  __shared__ float lg[20];
  __shared__ float probs[20];
  __shared__ float obuf[DD];
  __shared__ float hsum[HH];
  const int tid = threadIdx.x, lane = tid & 31, wave = tid >> 5;
  const int bl = blockIdx.x;
  if(tid < NSTAT) statv[tid] = stat[(size_t)bl * NSTAT + tid];
  if(tid < HH) hsum[tid] = 0.f;
  __syncthreads();
  #pragma unroll
  for(int j = 0; j < 8; j++){
    int n = tid + 256 * j;
    float a = g1pre[(size_t)bl * HF + n] + gb1[n];
    for(int g = 0; g < NSTAT; g++) a += statv[g] * g1red[(size_t)g * HF + n];
    h1[n] = 0.5f * a * (1.f + erff(a * 0.70710678f));      // exact gelu
  }
  __syncthreads();
  float pe[20];
  #pragma unroll
  for(int e = 0; e < 20; e++) pe[e] = 0.f;
  #pragma unroll
  for(int j = 0; j < 8; j++){
    int n = tid + 256 * j;
    float x = h1[n];
    #pragma unroll
    for(int e = 0; e < 20; e++) pe[e] += x * gW2[(size_t)n * 20 + e];
  }
  #pragma unroll
  for(int e = 0; e < 20; e++){
    float s = wred32(pe[e]);
    if(lane == 0) red[e * 8 + wave] = s;
  }
  __syncthreads();
  if(tid < 20){
    float s = 0.f;
    for(int w = 0; w < 8; w++) s += red[tid * 8 + w];
    lg[tid] = s + gb2[tid];
  }
  __syncthreads();
  if(tid < HH){
    int h = tid;
    float tau = log1pf(expf(log_tau[h])) + 0.5f;
    float mx = -1e30f;
    for(int p = 0; p < PP; p++) mx = fmaxf(mx, lg[h * PP + p] / tau);
    float den = 0.f, ex[PP], ev[PP], epsum = 0.f;
    for(int p = 0; p < PP; p++){ ex[p] = expf(lg[h * PP + p] / tau - mx); den += ex[p]; }
    for(int p = 0; p < PP; p++){ ev[p] = 0.2f / (1.f + expf(-eps_logit[h * PP + p])); epsum += ev[p]; }
    for(int p = 0; p < PP; p++) probs[h * PP + p] = (ex[p] / den) * (1.f - epsum) + ev[p];
  }
  __syncthreads();
  #pragma unroll
  for(int j = 0; j < 4; j++){
    int c = tid + 256 * j;
    int h = c >> 8;
    size_t gi = (size_t)bl * DD + c;
    float o = probs[h * PP + 0] * p0[gi] + probs[h * PP + 1] * p1[gi]
            + probs[h * PP + 2] * p2[gi] + probs[h * PP + 3] * p3[gi]
            + probs[h * PP + 4] * p4[gi];
    obuf[c] = o;
    atomicAdd(&hsum[h], o * o);
  }
  __syncthreads();
  #pragma unroll
  for(int j = 0; j < 4; j++){
    int c = tid + 256 * j;
    int h = c >> 8;
    float r = rsqrtf(hsum[h] * (1.f / DVv) + 1e-5f);
    orb[(size_t)bl * DD + c] = obuf[c] * r * norm_w[c & 255];
  }
}

// ---------------- chunkwise delta rule: S[256x128] resident in LDS ----------------
#define DELTA_SMEM (131072 + 3*4096 + 128 + 3*16384 + 2*8192 + 2*2048)
__launch_bounds__(256)
__global__ void delta_k(const float* __restrict__ q, const float* __restrict__ k,
                        const float* __restrict__ v, const float* __restrict__ beta,
                        float* __restrict__ od){
  extern __shared__ char smem[];
  float* S  = (float*)smem;                  // [256][128] f32 state slice
  float* G  = S + 256 * DVH;                 // [32][32] k k^T
  float* Lc = G + 1024;                      // [32][32] q k^T
  float* Tm = Lc + 1024;                     // [32][32] T
  float* bs = Tm + 1024;                     // [32] beta
  bf16* qb = (bf16*)(bs + 32);               // [32][256]
  bf16* kb = qb + 32 * DK;                   // [32][256]
  bf16* wb = kb + 32 * DK;                   // [32][256] stores -w
  bf16* vb = wb + 32 * DK;                   // [32][128]
  bf16* ub = vb + 32 * DVH;                  // [32][128]
  bf16* Tb = ub + 32 * DVH;                  // [32][32] T' = T*diag(beta)
  bf16* lb = Tb + 1024;                      // [32][32] incl-masked local

  const int tid = threadIdx.x, lane = tid & 31, wave = tid >> 5;
  const int half = lane >> 4, nn = lane & 15;
  const int bhh = blockIdx.x >> 1;
  const int b = bhh >> 2, h = bhh & 3;
  const int dvo = (blockIdx.x & 1) * DVH;

  for(int i = tid; i < 256 * DVH; i += 256) S[i] = 0.f;
  __syncthreads();

  #pragma unroll 1
  for(int ch = 0; ch < NCH; ch++){
    const int l0 = ch * CHK;
    // ---- stage 1: load + l2norm q,k ; load v-half
    if(tid < 32) bs[tid] = beta[((size_t)(b * LL + l0 + tid)) * HH + h];
    #pragma unroll 1
    for(int t = 0; t < 4; t++){
      int r = wave + 8 * t;
      size_t gq = ((size_t)(b * LL + l0 + r)) * DD + h * DK;
      float x[8], s = 0.f;
      #pragma unroll
      for(int j = 0; j < 8; j++){ x[j] = q[gq + lane + 32 * j]; s += x[j] * x[j]; }
      s = wred32(s);
      float rn = rsqrtf(s + 1e-6f);
      #pragma unroll
      for(int j = 0; j < 8; j++) qb[r * DK + lane + 32 * j] = f2bf(x[j] * rn);
      s = 0.f;
      #pragma unroll
      for(int j = 0; j < 8; j++){ x[j] = k[gq + lane + 32 * j]; s += x[j] * x[j]; }
      s = wred32(s);
      rn = rsqrtf(s + 1e-6f);
      #pragma unroll
      for(int j = 0; j < 8; j++) kb[r * DK + lane + 32 * j] = f2bf(x[j] * rn);
      #pragma unroll
      for(int j = 0; j < 4; j++) vb[r * DVH + lane + 32 * j] = f2bf(v[gq + dvo + lane + 32 * j]);
    }
    __syncthreads();
    // ---- stage 2: waves 0-3: G = k k^T ; waves 4-7: Lc = q k^T
    {
      int ti = (wave & 3) >> 1, tj = wave & 1;
      const bf16* src = (wave < 4) ? kb : qb;
      float* dst = (wave < 4) ? G : Lc;
      v8f g = {};
      #pragma unroll
      for(int ks = 0; ks < 8; ks++){
        v16bf a  = frag_a(src, DK, ti * 16, ks * 32, lane);
        v16bf bt = frag_bT(kb, DK, tj * 16, ks * 32, lane);
        g = wmma_bf(a, bt, g);
      }
      #pragma unroll
      for(int r = 0; r < 8; r++) dst[(ti * 16 + r + 8 * half) * 32 + tj * 16 + nn] = g[r];
    }
    __syncthreads();
    // ---- stage 3: T = (I + strict(diag(beta) G))^{-1} by fwd-subst (wave 0),
    //               T' = T diag(beta) -> Tb ; masked local -> lb (waves 1-7)
    if(wave == 0){
      int j = lane;
      for(int i = 0; i < 32; i++){
        float s = (i == j) ? 1.f : 0.f;
        float bi = bs[i];
        for(int m = 0; m < i; m++) s -= bi * G[i * 32 + m] * Tm[m * 32 + j];
        Tm[i * 32 + j] = s;
      }
      float bj = bs[j];
      for(int i = 0; i < 32; i++) Tb[i * 32 + j] = f2bf(Tm[i * 32 + j] * bj);
    } else {
      for(int ii = tid - 32; ii < 1024; ii += 224){
        int i = ii >> 5, j = ii & 31;
        lb[ii] = (j <= i) ? f2bf(Lc[ii]) : f2bf(0.f);
      }
    }
    __syncthreads();
    // ---- stage 4: wb = -(T' @ k) ; uacc = T' @ v ; oacc = q @ S
    v8f uacc[2] = {}, oacc[2] = {};
    #pragma unroll 1
    for(int t = 0; t < 4; t++){
      int idx = wave * 4 + t, ti = idx >> 4, tj = idx & 15;
      v16bf a   = frag_a(Tb, 32, ti * 16, 0, lane);
      v16bf bfr = frag_b(kb, DK, 0, tj * 16, lane);
      v8f wv = {};
      wv = wmma_bf(a, bfr, wv);
      #pragma unroll
      for(int r = 0; r < 8; r++) wb[(ti * 16 + r + 8 * half) * DK + tj * 16 + nn] = f2bf(-wv[r]);
    }
    #pragma unroll 1
    for(int t = 0; t < 2; t++){
      int idx = wave * 2 + t, ti = idx >> 3, tj = idx & 7;
      v16bf a   = frag_a(Tb, 32, ti * 16, 0, lane);
      v16bf bfr = frag_b(vb, DVH, 0, tj * 16, lane);
      uacc[t] = wmma_bf(a, bfr, uacc[t]);
      #pragma unroll 1
      for(int ks = 0; ks < 8; ks++){
        v16bf aq = frag_a(qb, DK, ti * 16, ks * 32, lane);
        v16bf bS = frag_b_f32(S, DVH, ks * 32, tj * 16, lane);
        oacc[t] = wmma_bf(aq, bS, oacc[t]);
      }
    }
    __syncthreads();
    // ---- stage 5: uacc += (-w) @ S ; write u_i
    #pragma unroll 1
    for(int t = 0; t < 2; t++){
      int idx = wave * 2 + t, ti = idx >> 3, tj = idx & 7;
      #pragma unroll 1
      for(int ks = 0; ks < 8; ks++){
        v16bf aw = frag_a(wb, DK, ti * 16, ks * 32, lane);
        v16bf bS = frag_b_f32(S, DVH, ks * 32, tj * 16, lane);
        uacc[t] = wmma_bf(aw, bS, uacc[t]);
      }
      #pragma unroll
      for(int r = 0; r < 8; r++) ub[(ti * 16 + r + 8 * half) * DVH + tj * 16 + nn] = f2bf(uacc[t][r]);
    }
    __syncthreads();
    // ---- stage 6: o = oacc + local @ u -> global ; S += k^T @ u
    #pragma unroll 1
    for(int t = 0; t < 2; t++){
      int idx = wave * 2 + t, ti = idx >> 3, tj = idx & 7;
      v16bf al = frag_a(lb, 32, ti * 16, 0, lane);
      v16bf bu = frag_b(ub, DVH, 0, tj * 16, lane);
      oacc[t] = wmma_bf(al, bu, oacc[t]);
      #pragma unroll
      for(int r = 0; r < 8; r++){
        int row = ti * 16 + r + 8 * half, col = tj * 16 + nn;
        od[((size_t)(b * LL + l0 + row)) * DD + h * DVv + dvo + col] = oacc[t][r];
      }
    }
    #pragma unroll 1
    for(int t = 0; t < 16; t++){
      int idx = wave * 16 + t, mi = idx >> 3, ni = idx & 7;
      v8f c;
      #pragma unroll
      for(int r = 0; r < 8; r++) c[r] = S[(mi * 16 + r + 8 * half) * DVH + ni * 16 + nn];
      v16bf aT = frag_aT(kb, DK, mi * 16, 0, lane);
      v16bf bu = frag_b(ub, DVH, 0, ni * 16, lane);
      c = wmma_bf(aT, bu, c);
      #pragma unroll
      for(int r = 0; r < 8; r++) S[(mi * 16 + r + 8 * half) * DVH + ni * 16 + nn] = c[r];
    }
    __syncthreads();
  }
}

// ---------------- host orchestration ----------------
extern "C" void kernel_launch(void* const* d_in, const int* in_sizes, int n_in,
                              void* d_out, int out_size, void* d_ws, size_t ws_size,
                              hipStream_t stream){
  (void)in_sizes; (void)n_in; (void)out_size; (void)ws_size;
  const float* hs      = (const float*)d_in[0];
  const float* Wq      = (const float*)d_in[1];
  const float* Wk      = (const float*)d_in[2];
  const float* Wv      = (const float*)d_in[3];
  const float* conv_q  = (const float*)d_in[4];
  const float* conv_k  = (const float*)d_in[5];
  const float* conv_v  = (const float*)d_in[6];
  const float* Wb      = (const float*)d_in[7];
  const float* fir_s   = (const float*)d_in[8];
  const float* fir_m   = (const float*)d_in[9];
  const float* fir_l   = (const float*)d_in[10];
  const float* gW1     = (const float*)d_in[11];
  const float* gb1     = (const float*)d_in[12];
  const float* gW2     = (const float*)d_in[13];
  const float* gb2     = (const float*)d_in[14];
  const float* log_tau = (const float*)d_in[15];
  const float* epsl    = (const float*)d_in[16];
  const float* norm_w  = (const float*)d_in[17];
  const float* Wo      = (const float*)d_in[18];

  char* ws = (char*)d_ws;
  const size_t MB = 1ull << 20;
  float* qpre  = (float*)(ws + 0 * MB);     // reused as fir_s output
  float* kpre  = (float*)(ws + 16 * MB);    // reused as fir_m output
  float* vpre  = (float*)(ws + 32 * MB);    // reused as fir_l output
  float* qf    = (float*)(ws + 48 * MB);
  float* kf    = (float*)(ws + 64 * MB);
  float* vf    = (float*)(ws + 80 * MB);
  float* dlt   = (float*)(ws + 96 * MB);
  float* orb   = (float*)(ws + 112 * MB);
  float* g1pre = (float*)(ws + 128 * MB);
  float* betab = (float*)(ws + 160 * MB);
  float* statb = (float*)(ws + 161 * MB);
  float* g1red = (float*)(ws + 163 * MB);

  dim3 blk(256);
  // q/k/v projections + hs-part of gate layer 1 (all bf16 WMMA, f32 accum)
  gemm_wmma<<<dim3(8, 64),  blk, 0, stream>>>(hs, Wq,  qpre,  BL, DD, DD);
  gemm_wmma<<<dim3(8, 64),  blk, 0, stream>>>(hs, Wk,  kpre,  BL, DD, DD);
  gemm_wmma<<<dim3(8, 64),  blk, 0, stream>>>(hs, Wv,  vpre,  BL, DD, DD);
  gemm_wmma<<<dim3(16, 64), blk, 0, stream>>>(hs, gW1, g1pre, BL, HF, DD);
  // rank-1 stats-weight reduction + beta
  reduce_gw1_k<<<640, blk, 0, stream>>>(gW1, g1red);
  beta_k<<<64, blk, 0, stream>>>(hs, Wb, betab);
  // short causal convs + SiLU
  dwconv_k<<<4096, blk, 0, stream>>>(qpre, conv_q, qf, 4, 1);
  dwconv_k<<<4096, blk, 0, stream>>>(kpre, conv_k, kf, 4, 1);
  dwconv_k<<<4096, blk, 0, stream>>>(vpre, conv_v, vf, 4, 1);
  // chunkwise delta rule: 16 WGs = (b,h,dv-half), S lives in LDS
  delta_k<<<16, blk, DELTA_SMEM, stream>>>(qf, kf, vf, betab, dlt);
  // FIR paths (reuse pre-buffers)
  dwconv_k<<<4096, blk, 0, stream>>>(vf, fir_s, qpre, 3,  0);
  dwconv_k<<<4096, blk, 0, stream>>>(vf, fir_m, kpre, 15, 0);
  dwconv_k<<<4096, blk, 0, stream>>>(vf, fir_l, vpre, 64, 0);
  // stats -> gate MLP -> mix -> RMSNorm
  stats_k<<<4096, blk, 0, stream>>>(qpre, kpre, vpre, dlt, vf, statb);
  gate_k<<<4096, blk, 0, stream>>>(g1pre, statb, g1red, gb1, gW2, gb2, log_tau, epsl, norm_w,
                                   qpre, kpre, vpre, dlt, vf, orb);
  // output projection
  gemm_wmma<<<dim3(8, 64), blk, 0, stream>>>(orb, Wo, (float*)d_out, BL, DD, DD);
}
